// DenseGCN3Layer_83124797046811
// MI455X (gfx1250) — matrix operand order repaired and checked by reading, not compiled, not run on hardware
//
#include <hip/hip_runtime.h>
#include <hip/hip_bf16.h>

typedef __attribute__((ext_vector_type(16))) __bf16 v16bf;
typedef __attribute__((ext_vector_type(8)))  float  v8f;

// ---------------------------------------------------------------------------
// Degree count: deg[dst[e]] += 1
// ---------------------------------------------------------------------------
__global__ void deg_count_kernel(const int* __restrict__ dst,
                                 float* __restrict__ deg, int E) {
  int e = blockIdx.x * blockDim.x + threadIdx.x;
  if (e < E) atomicAdd(&deg[dst[e]], 1.0f);
}

// deg -> rsqrt(deg + 1) in place
__global__ void dinv_kernel(float* __restrict__ d, int N) {
  int i = blockIdx.x * blockDim.x + threadIdx.x;
  if (i < N) d[i] = rsqrtf(d[i] + 1.0f);
}

// ---------------------------------------------------------------------------
// C[M,N] = A[M,K] @ B[K,N] (+ bias[n]) (+= existing C if ACCUM)
// fp32 inputs converted to bf16 on load; fp32 accumulate via
// v_wmma_f32_16x16x32_bf16. One wave computes a full 16 x (NT*16) strip:
// the A fragment is loaded once per k-step and reused across NT WMMAs.
// Requires: M % 16 == 0, N == NT*16, K % 32 == 0 (all true here).
// ---------------------------------------------------------------------------
template <int NT, bool ACCUM>
__global__ void gemm_wmma_bf16(const float* __restrict__ A,
                               const float* __restrict__ B,
                               const float* __restrict__ bias,
                               float* __restrict__ C,
                               int M, int N, int K) {
  const int wave = (blockIdx.x * blockDim.x + threadIdx.x) >> 5;
  const int lane = threadIdx.x & 31;
  if (wave * 16 >= M) return;            // uniform per-wave exit; EXEC stays full

  const int half = lane >> 4;            // 0: lanes 0-15, 1: lanes 16-31
  const int lidx = lane & 15;
  const int row  = wave * 16 + lidx;     // A row for this lane

  v8f acc[NT];
  #pragma unroll
  for (int t = 0; t < NT; ++t) acc[t] = (v8f){};

  for (int k0 = 0; k0 < K; k0 += 32) {
    // A fragment: 16-bit A 16x32 layout (ISA 7.12.2); loaded once, used NT times
    v16bf a;
    const float* Arow = A + (long long)row * K + k0;
    #pragma unroll
    for (int i = 0; i < 8; ++i) a[i]     = (__bf16)Arow[half * 8 + i];       // K = h*8 + i
    #pragma unroll
    for (int i = 0; i < 8; ++i) a[8 + i] = (__bf16)Arow[16 + half * 8 + i];  // K = 16 + h*8 + i

    #pragma unroll
    for (int t = 0; t < NT; ++t) {
      // B fragment: 32x16, lane holds column t*16+lidx, K = half*8 + i (i<8), 16 + half*8 + (i-8) (i>=8)
      v16bf b;
      const float* Bcol = B + (long long)(k0 + half * 8) * N + t * 16 + lidx;
      #pragma unroll
      for (int i = 0; i < 16; ++i) b[i] = (__bf16)Bcol[(long long)(i + (i >> 3) * 8) * N];

      acc[t] = __builtin_amdgcn_wmma_f32_16x16x32_bf16(
          /*neg_a=*/false, a, /*neg_b=*/false, b,
          /*c_mod=*/(short)0, acc[t], /*reuse_a=*/false, /*reuse_b=*/false);
    }
  }

  #pragma unroll
  for (int t = 0; t < NT; ++t) {
    const int col  = t * 16 + lidx;
    const float bv = bias ? bias[col] : 0.0f;
    float* Cp = C + (long long)(wave * 16) * N + col;
    #pragma unroll
    for (int r = 0; r < 8; ++r) {        // C/D: VGPR r -> M = r + 8*half
      long long off = (long long)(r + half * 8) * N;
      float v = acc[t][r] + bv;
      if (ACCUM) v += Cp[off];
      Cp[off] = v;
    }
  }
}

// ---------------------------------------------------------------------------
// Edge scatter: agg[dst[e], f] += dinv[src[e]]*dinv[dst[e]] * h[src[e], f]
// One thread per (edge, feature); F = 1<<fshift so gathers/atomics are
// contiguous in f (coalesced, L2-resident).
// ---------------------------------------------------------------------------
__global__ void edge_scatter_kernel(const int* __restrict__ src,
                                    const int* __restrict__ dst,
                                    const float* __restrict__ dinv,
                                    const float* __restrict__ h,
                                    float* __restrict__ agg,
                                    long long total, int fshift) {
  long long idx = (long long)blockIdx.x * blockDim.x + threadIdx.x;
  if (idx >= total) return;
  const int f       = (int)(idx & ((1 << fshift) - 1));
  const long long e = idx >> fshift;
  const int s = src[e];
  const int d = dst[e];
  const float norm = dinv[s] * dinv[d];
  atomicAdd(&agg[((long long)d << fshift) + f],
            norm * h[((long long)s << fshift) + f]);
}

// ---------------------------------------------------------------------------
// out[i,f] = relu(agg[i,f] + dinv[i]^2 * h[i,f] + b[f] + (base ? base[i,f] : 0))
// ---------------------------------------------------------------------------
__global__ void finalize_kernel(const float* __restrict__ agg,
                                const float* __restrict__ h,
                                const float* __restrict__ base,
                                const float* __restrict__ dinv,
                                const float* __restrict__ b,
                                float* __restrict__ out,
                                long long total, int fshift) {
  long long idx = (long long)blockIdx.x * blockDim.x + threadIdx.x;
  if (idx >= total) return;
  const int f       = (int)(idx & ((1 << fshift) - 1));
  const long long i = idx >> fshift;
  const float di = dinv[i];
  float v = agg[idx] + di * di * h[idx] + b[f];
  if (base) v += base[idx];
  out[idx] = fmaxf(v, 0.0f);
}

// ---------------------------------------------------------------------------
// out[i] = sigmoid(dot(x3[i, 0:16], Wout) + bout)
// ---------------------------------------------------------------------------
__global__ void head_kernel(const float* __restrict__ x3,
                            const float* __restrict__ Wout,
                            const float* __restrict__ bout,
                            float* __restrict__ out, int N) {
  int i = blockIdx.x * blockDim.x + threadIdx.x;
  if (i >= N) return;
  float s = bout[0];
  const float* row = x3 + (long long)i * 16;
  #pragma unroll
  for (int f = 0; f < 16; ++f) s += row[f] * Wout[f];
  out[i] = 1.0f / (1.0f + __expf(-s));
}

// ---------------------------------------------------------------------------
extern "C" void kernel_launch(void* const* d_in, const int* in_sizes, int n_in,
                              void* d_out, int out_size, void* d_ws, size_t ws_size,
                              hipStream_t stream) {
  const float* x    = (const float*)d_in[0];
  const int*   ei   = (const int*)d_in[1];
  const float* W1   = (const float*)d_in[2];
  const float* b1   = (const float*)d_in[3];
  const float* W2   = (const float*)d_in[4];
  const float* b2   = (const float*)d_in[5];
  const float* W3   = (const float*)d_in[6];
  const float* b3   = (const float*)d_in[7];
  const float* Ws02 = (const float*)d_in[8];
  const float* bs02 = (const float*)d_in[9];
  const float* Ws03 = (const float*)d_in[10];
  const float* bs03 = (const float*)d_in[11];
  const float* Ws13 = (const float*)d_in[12];
  const float* bs13 = (const float*)d_in[13];
  const float* Wout = (const float*)d_in[14];
  const float* bout = (const float*)d_in[15];

  const int N = in_sizes[0] / 256;   // 100000
  const int E = in_sizes[1] / 2;     // 3200000
  const int* src = ei;               // edge_index row 0
  const int* dst = ei + E;           // edge_index row 1
  float* out = (float*)d_out;

  // ---- workspace layout (floats) ----
  float* ws    = (float*)d_ws;
  float* dinv  = ws;  ws += N;
  float* h1    = ws;  ws += (size_t)N * 64;
  float* agg1  = ws;  ws += (size_t)N * 64;
  float* x1    = ws;  ws += (size_t)N * 64;
  float* h2    = ws;  ws += (size_t)N * 32;
  float* base2 = ws;  ws += (size_t)N * 32;
  float* agg2  = ws;  ws += (size_t)N * 32;
  float* x2    = ws;  ws += (size_t)N * 32;
  float* h3    = ws;  ws += (size_t)N * 16;
  float* base3 = ws;  ws += (size_t)N * 16;
  float* agg3  = ws;  ws += (size_t)N * 16;
  float* x3    = ws;  ws += (size_t)N * 16;

  const int TB = 256;  // 8 wave32s per block

  auto gemm = [&](const float* A, const float* B, const float* bias, float* C,
                  int M, int Ncols, int K, bool accum) {
    long long waves  = (long long)(M / 16);     // one wave per 16-row strip
    int       blocks = (int)((waves + 7) / 8);
    if (accum) {
      // only used for Ncols == 16
      gemm_wmma_bf16<1, true ><<<blocks, TB, 0, stream>>>(A, B, bias, C, M, Ncols, K);
    } else if (Ncols == 64) {
      gemm_wmma_bf16<4, false><<<blocks, TB, 0, stream>>>(A, B, bias, C, M, Ncols, K);
    } else if (Ncols == 32) {
      gemm_wmma_bf16<2, false><<<blocks, TB, 0, stream>>>(A, B, bias, C, M, Ncols, K);
    } else {
      gemm_wmma_bf16<1, false><<<blocks, TB, 0, stream>>>(A, B, bias, C, M, Ncols, K);
    }
  };
  auto scatter = [&](const float* h, float* agg, int fshift) {
    long long total  = (long long)E << fshift;
    int       blocks = (int)((total + TB - 1) / TB);
    edge_scatter_kernel<<<blocks, TB, 0, stream>>>(src, dst, dinv, h, agg, total, fshift);
  };
  auto finalize = [&](const float* agg, const float* h, const float* base,
                      const float* b, float* xo, int fshift) {
    long long total  = (long long)N << fshift;
    int       blocks = (int)((total + TB - 1) / TB);
    finalize_kernel<<<blocks, TB, 0, stream>>>(agg, h, base, dinv, b, xo, total, fshift);
  };

  // ---- degree / normalization ----
  hipMemsetAsync(dinv, 0, (size_t)N * sizeof(float), stream);
  deg_count_kernel<<<(E + TB - 1) / TB, TB, 0, stream>>>(dst, dinv, E);
  dinv_kernel<<<(N + TB - 1) / TB, TB, 0, stream>>>(dinv, N);

  // ---- layer 1: x1 = relu(gcn(x, W1, b1)) ----
  gemm(x, W1, nullptr, h1, N, 64, 256, false);
  hipMemsetAsync(agg1, 0, (size_t)N * 64 * sizeof(float), stream);
  scatter(h1, agg1, 6);
  finalize(agg1, h1, nullptr, b1, x1, 6);

  // ---- layer 2: x2 = relu(gcn(x1, W2, b2) + x@Ws02 + bs02) ----
  gemm(x1, W2, nullptr, h2, N, 32, 64, false);
  gemm(x, Ws02, bs02, base2, N, 32, 256, false);
  hipMemsetAsync(agg2, 0, (size_t)N * 32 * sizeof(float), stream);
  scatter(h2, agg2, 5);
  finalize(agg2, h2, base2, b2, x2, 5);

  // ---- layer 3: x3 = relu(gcn(x2, W3, b3) + x@Ws03 + bs03 + x1@Ws13 + bs13) ----
  gemm(x2, W3, nullptr, h3, N, 16, 32, false);
  gemm(x, Ws03, bs03, base3, N, 16, 256, false);
  gemm(x1, Ws13, bs13, base3, N, 16, 64, true);   // accumulate skip into base3
  hipMemsetAsync(agg3, 0, (size_t)N * 16 * sizeof(float), stream);
  scatter(h3, agg3, 4);
  finalize(agg3, h3, base3, b3, x3, 4);

  // ---- head: sigmoid(x3 @ Wout + bout) ----
  head_kernel<<<(N + TB - 1) / TB, TB, 0, stream>>>(x3, Wout, bout, out, N);
}
